// Net_53687091200003
// MI455X (gfx1250) — matrix operand order repaired and checked by reading, not compiled, not run on hardware
//
#include <hip/hip_runtime.h>

typedef __attribute__((ext_vector_type(16))) _Float16 v16h;
typedef __attribute__((ext_vector_type(8)))  _Float16 h8;
typedef __attribute__((ext_vector_type(2)))  _Float16 h2;
typedef __attribute__((ext_vector_type(8)))  float    v8f;

#define TT   256
#define DD   23
#define NH1  100
#define NH2  100
#define NO   2
#define BETA 0.95f
#define THR  1.0f

#define XA_STRIDE 40    // halves; row stride for x tile [16][32] (+8 pad, bank rotation)
#define SP_STRIDE 136   // halves; row stride for spike tiles [16][128] (+8 pad)

// Build a WMMA B-fragment (32x16, f16) for W stored row-major [Nv][Kv]:
// B[k][n] = W[n][k], zero-padded outside valid range.
// Layout: half h of lane ln holds K = kbase + 8*(ln>=16) + (h<8 ? h : h+8), N = ntile*16 + (ln&15).
__device__ inline v16h make_bfrag(const float* __restrict__ W, int Nv, int Kv,
                                  int ntile, int kbase, int lane) {
    union { v16h v; _Float16 e[16]; } f;
    const int n  = ntile * 16 + (lane & 15);
    const int kb = kbase + ((lane & 16) ? 8 : 0);
#pragma unroll
    for (int h = 0; h < 16; ++h) {
        const int k = kb + (h < 8 ? h : h + 8);
        const float w = (n < Nv && k < Kv) ? W[n * Kv + k] : 0.0f;
        f.e[h] = (_Float16)w;
    }
    return f.v;
}

// Load a WMMA A-fragment (16x32, f16) for K-chunk `kc` from an LDS tile
// stored row-major [16][stride halves]. Two 16B ds_read_b128 per lane.
// Layout: lane<16: halves0-7=K(kc*32+0..7), halves8-15=K(+16..23); lane>=16: +8 / +24.
__device__ inline v16h load_afrag(const _Float16* lds, int stride, int kc, int lane) {
    const int row  = lane & 15;
    const int koff = kc * 32 + ((lane & 16) ? 8 : 0);
    union { v16h v; h8 h[2]; } f;
    f.h[0] = *(const h8*)(lds + row * stride + koff);
    f.h[1] = *(const h8*)(lds + row * stride + koff + 16);
    return f.v;
}

// Scatter a spike C-tile (16x16 f32 accumulator layout) into an LDS f16 tile.
// C layout: lane ln holds column n=(ln&15)+16*ntile; element r holds row m=r+8*(ln>=16).
__device__ inline void store_spk(_Float16* lds, const v8f& s, int ntile, int lane) {
    const int n     = ntile * 16 + (lane & 15);
    const int mbase = (lane & 16) ? 8 : 0;
#pragma unroll
    for (int r = 0; r < 8; ++r)
        lds[(mbase + r) * SP_STRIDE + n] = (_Float16)s[r];
}

__global__ __launch_bounds__(256) void snn_lif3_kernel(
    const float* __restrict__ x,
    const float* __restrict__ W1, const float* __restrict__ b1,
    const float* __restrict__ W2, const float* __restrict__ b2,
    const float* __restrict__ W3, const float* __restrict__ b3,
    float* __restrict__ out, int B)
{
    __shared__ _Float16 xa [16 * XA_STRIDE];
    __shared__ _Float16 sp1[16 * SP_STRIDE];
    __shared__ _Float16 sp2[16 * SP_STRIDE];

    const int tid  = threadIdx.x;
    const int wave = tid >> 5;
    const int lane = tid & 31;
    const int base = blockIdx.x * 16;                 // batch row base of this block
    const size_t memOff = (size_t)B * TT * NO;        // mem_rec offset in d_out

    // Zero spike tiles once (covers the n=112..127 pad columns never rewritten).
    for (int i = tid; i < 16 * SP_STRIDE; i += 256) {
        sp1[i] = (_Float16)0.0f;
        sp2[i] = (_Float16)0.0f;
    }

    // ---- persistent per-wave weight fragments & bias ----
    v16h w1f;        // waves 0..6: layer-1 B-frag (K=32 covering D=23)
    v16h wf[4];      // waves 0..6: layer-2 B-frags (K=128 covering 100); wave 7: layer-3
    float bA = 0.0f, bB = 0.0f;

    if (wave < 7) {
        w1f = make_bfrag(W1, NH1, DD, wave, 0, lane);
#pragma unroll
        for (int c = 0; c < 4; ++c) wf[c] = make_bfrag(W2, NH2, NH1, wave, 32 * c, lane);
        const int n = wave * 16 + (lane & 15);
        bA = (n < NH1) ? b1[n] : 0.0f;
        bB = (n < NH2) ? b2[n] : 0.0f;
    } else {
        w1f = v16h{};
#pragma unroll
        for (int c = 0; c < 4; ++c) wf[c] = make_bfrag(W3, NO, NH2, 0, 32 * c, lane);
        const int n = lane & 15;
        bB = (n < NO) ? b3[n] : 0.0f;
    }

    // LIF state in accumulator layout. Waves 0..6: A=layer1, B=layer2. Wave 7: B=layer3.
    v8f memA = {}, memB = {}, sA = {}, sB = {};

    __syncthreads();

    for (int t = 0; t < TT; ++t) {
        // ---- stage x[base..base+15][t][0..22] -> f16, zero-padded to K=32 ----
        {
            const int h0 = tid * 2;              // 256 threads x 2 halves = 16*32
            const int m  = h0 >> 5;
            const int k  = h0 & 31;
            const float* xr = x + ((size_t)(base + m) * TT + t) * DD;
            h2 pk;
            pk[0] = (_Float16)((k     < DD) ? xr[k]     : 0.0f);
            pk[1] = (_Float16)((k + 1 < DD) ? xr[k + 1] : 0.0f);
            *(h2*)(xa + m * XA_STRIDE + k) = pk;
        }
        __syncthreads();

        // ---- layer 1: mem1 = beta*mem1 + x@W1^T + b1 - spk1_prev ----
        if (wave < 7) {
            v8f acc;
#pragma unroll
            for (int i = 0; i < 8; ++i) acc[i] = BETA * memA[i] + bA - sA[i] * THR;
            const v16h a = load_afrag(xa, XA_STRIDE, 0, lane);
            acc = __builtin_amdgcn_wmma_f32_16x16x32_f16(false, a, false, w1f,
                                                         (short)0, acc, false, false);
            memA = acc;
#pragma unroll
            for (int i = 0; i < 8; ++i) sA[i] = (acc[i] > THR) ? 1.0f : 0.0f;
            store_spk(sp1, sA, wave, lane);
        }
        __syncthreads();

        // ---- layer 2: mem2 = beta*mem2 + spk1@W2^T + b2 - spk2_prev ----
        if (wave < 7) {
            v8f acc;
#pragma unroll
            for (int i = 0; i < 8; ++i) acc[i] = BETA * memB[i] + bB - sB[i] * THR;
#pragma unroll
            for (int c = 0; c < 4; ++c) {
                const v16h a = load_afrag(sp1, SP_STRIDE, c, lane);
                acc = __builtin_amdgcn_wmma_f32_16x16x32_f16(false, a, false, wf[c],
                                                             (short)0, acc, false, false);
            }
            memB = acc;
#pragma unroll
            for (int i = 0; i < 8; ++i) sB[i] = (acc[i] > THR) ? 1.0f : 0.0f;
            store_spk(sp2, sB, wave, lane);
        }
        __syncthreads();

        // ---- layer 3 (wave 7): mem3 = beta*mem3 + spk2@W3^T + b3 - spk3_prev ----
        if (wave == 7) {
            v8f acc;
#pragma unroll
            for (int i = 0; i < 8; ++i) acc[i] = BETA * memB[i] + bB - sB[i] * THR;
#pragma unroll
            for (int c = 0; c < 4; ++c) {
                const v16h a = load_afrag(sp2, SP_STRIDE, c, lane);
                acc = __builtin_amdgcn_wmma_f32_16x16x32_f16(false, a, false, wf[c],
                                                             (short)0, acc, false, false);
            }
            memB = acc;
#pragma unroll
            for (int i = 0; i < 8; ++i) sB[i] = (acc[i] > THR) ? 1.0f : 0.0f;

            // Emit spk3/mem3 for the 2 valid output neurons.
            const int n = lane & 15;
            if (n < NO) {
                const int mb = base + ((lane & 16) ? 8 : 0);
#pragma unroll
                for (int r = 0; r < 8; ++r) {
                    const size_t idx = ((size_t)(mb + r) * TT + t) * NO + n;
                    out[idx]          = sB[r];
                    out[memOff + idx] = memB[r];
                }
            }
        }
        // next-iteration x staging is ordered by the first barrier of the next step;
        // wave 7's sp2 reads complete before it can pass that barrier, and sp2 is
        // only rewritten after the second barrier of the next step.
    }
}

extern "C" void kernel_launch(void* const* d_in, const int* in_sizes, int n_in,
                              void* d_out, int out_size, void* d_ws, size_t ws_size,
                              hipStream_t stream) {
    const float* x  = (const float*)d_in[0];
    const float* W1 = (const float*)d_in[1];
    const float* b1 = (const float*)d_in[2];
    const float* W2 = (const float*)d_in[3];
    const float* b2 = (const float*)d_in[4];
    const float* W3 = (const float*)d_in[5];
    const float* b3 = (const float*)d_in[6];
    float* out = (float*)d_out;

    const int B = in_sizes[0] / (TT * DD);   // 16384
    const int grid = B / 16;                 // 16 batch rows per block

    snn_lif3_kernel<<<grid, 256, 0, stream>>>(x, W1, b1, W2, b2, W3, b3, out, B);
}
